// RIM_51728586113606
// MI455X (gfx1250) — compile-verified, matrix operand order checked by reading, not run on hardware
//
#include <hip/hip_runtime.h>
#include <hip/hip_bf16.h>

// ---------------------------------------------------------------------------
// Fused implementation for MI455X (gfx1250, wave32, WMMA).
// All (B,256) GEMMs run as v_wmma_f32_16x16x32_bf16 with f32 accumulation.
// 64 rows per workgroup: each wave holds 4 row-tiles of accumulators so every
// B fragment read from L2 feeds four WMMAs (~64 FLOP per L2 byte).
// LDS buffers are lifetime-aliased (bn | g+b, a | h) to fit 130KB so two
// workgroups still co-reside per WGP.
// ---------------------------------------------------------------------------

#define BB   10000
#define FF   512
#define EE   256
#define LDIM 128
#define OUTD 64
#define ROWS 64                 // rows per workgroup (4 x 16-row WMMA tiles)

typedef __attribute__((ext_vector_type(16))) __bf16 v16bf;
typedef __attribute__((ext_vector_type(8)))  __bf16 v8bf;
typedef __attribute__((ext_vector_type(8)))  float  v8f;

struct P {
  // inputs (f32)
  const float *x, *eps1, *eps2, *bn;
  const float *geW1, *geb1, *geW2, *geb2;
  const float *prencW,*prencb,*prgW,*prgb,*prbW,*prbb,*prmeanW,*prmeanb,*prstdW,*prstdb;
  const float *poencW,*poencb,*pogW,*pogb,*pobW,*pobb,*pomeanW,*pomeanb,*postdW,*postdb;
  const float *prmijW,*prmijb,*pomapxW,*pomapxb,*posapxW,*posapxb;
  const float *outW1,*outb1,*outW2,*outb2;
  // workspace (bf16 transposed weights [N][K], f32 vectors)
  __bf16 *W1t,*W2t,*prencT,*prgT,*prbT,*poencT,*pogT,*pobT;
  float *xvec, *Hg;
  float *out; // d_out: [0..63]=x_out, [64]=kl_g, [65]=kl_b, [66..]=alpha_bar
};

__device__ __forceinline__ float sp_(float v)   { return v > 20.f ? v : log1pf(expf(v)); }
__device__ __forceinline__ float sigm_(float v) { return 1.f / (1.f + expf(-v)); }

// Load one 16-bit WMMA fragment half-pair: lane holds 8 contiguous bf16 at p
// and 8 more at p+16 (matches ISA 16-bit A/B layout with kh = (lane>>4)*8).
__device__ __forceinline__ v16bf ldfrag(const __bf16* p) {
  const v8bf* q = (const v8bf*)p;
  v8bf lo = q[0];
  v8bf hi = q[2];
  v16bf r;
#pragma unroll
  for (int i = 0; i < 8; ++i) { r[i] = lo[i]; r[i + 8] = hi[i]; }
  return r;
}

// One wave computes a 64xK @ Kx64 slab as four 16-row WMMA tiles that share
// every B fragment. A tile (64 rows) in LDS, B = bf16 transposed weights
// [256][K] in L2-resident workspace. Output -> LDS bf16, f32 bias + opt relu.
__device__ __forceinline__ void wave_gemm(const __bf16* __restrict__ A,      // LDS [64][K]
                                          const __bf16* __restrict__ Wt,     // [256][K]
                                          const float*  __restrict__ bias,   // [256] f32
                                          bool relu,
                                          __bf16* __restrict__ Out,          // LDS [64][256]
                                          int K) {
  const int lane = threadIdx.x & 31;
  const int wave = threadIdx.x >> 5;
  const int col0 = wave * 64;
  const int r16  = lane & 15;
  const int kh   = (lane >> 4) * 8;

  v8f acc[4][4];
#pragma unroll
  for (int rt = 0; rt < 4; ++rt)
#pragma unroll
    for (int ct = 0; ct < 4; ++ct) acc[rt][ct] = v8f{};

  for (int k0 = 0; k0 < K; k0 += 32) {
    v16bf a[4];
#pragma unroll
    for (int rt = 0; rt < 4; ++rt)
      a[rt] = ldfrag(A + (size_t)(rt * 16 + r16) * K + k0 + kh);
#pragma unroll
    for (int ct = 0; ct < 4; ++ct) {
      const int col = col0 + ct * 16 + r16;
      v16bf b = ldfrag(Wt + (size_t)col * K + k0 + kh);
#pragma unroll
      for (int rt = 0; rt < 4; ++rt)
        acc[rt][ct] = __builtin_amdgcn_wmma_f32_16x16x32_bf16(
            false, a[rt], false, b, (short)0, acc[rt][ct], false, false);
    }
  }
  const int mb = (lane >> 4) * 8;  // C/D layout: lanes 16-31 hold rows 8..15
#pragma unroll
  for (int rt = 0; rt < 4; ++rt) {
#pragma unroll
    for (int ct = 0; ct < 4; ++ct) {
      const int n  = col0 + ct * 16 + r16;
      const float bv = bias[n];
#pragma unroll
      for (int r = 0; r < 8; ++r) {
        float v = acc[rt][ct][r] + bv;
        if (relu) v = fmaxf(v, 0.f);
        Out[(size_t)(rt * 16 + mb + r) * EE + n] = (__bf16)v;
      }
    }
  }
}

// ---- prep: transpose-convert one f32 [K][N] matrix to bf16 [N][K] ----------
__global__ void k_conv(const float* __restrict__ src, __bf16* __restrict__ dst,
                       int K, int N) {
  int total = K * N;
  for (int idx = blockIdx.x * blockDim.x + threadIdx.x; idx < total;
       idx += gridDim.x * blockDim.x) {
    int n = idx / K, k = idx - n * K;
    dst[idx] = (__bf16)src[(size_t)k * N + n];
  }
}

// ---- prep: xvec = x @ W1[512:,:] + ge_b1 ; zero Hg and kl accumulators -----
__global__ __launch_bounds__(512) void k_prep(P p) {
  int t = threadIdx.x;
  if (t < FF) p.Hg[t] = 0.f;
  if (t == 0) { p.out[64] = 0.f; p.out[65] = 0.f; }
  if (t < EE) {
    float acc = p.geb1[t];
    for (int k = 0; k < FF; ++k)
      acc += p.x[k] * p.geW1[(size_t)(FF + k) * EE + t];
    p.xvec[t] = acc;
  }
}

// ---- main fused kernel: 64 rows per block, 4 waves ------------------------
// LDS aliasing (lifetimes separated by __syncthreads):
//   [0,64K)    sBN (stage0..GEMM1)   ALIASES  sG[0,32K) + sB[32K,64K) (later)
//   [64K,96K)  sA  (GEMM1..GEMM2)    ALIASES  sH (prenc onward)
//   [96K,128K) sE  (GEMM2..poenc)
//   [128K,130K) sS per-row scalars
__global__ __launch_bounds__(128) void k_main(P p) {
  __shared__ __align__(16) char smem[ROWS * FF * 2 + 2 * ROWS * EE * 2 + ROWS * 8 * 4];
  __bf16* sBN = (__bf16*)smem;                          // 64K
  __bf16* sG  = (__bf16*)smem;                          // 32K (aliases sBN lo)
  __bf16* sB  = (__bf16*)(smem + ROWS * EE * 2);        // 32K (aliases sBN hi)
  __bf16* sA  = (__bf16*)(smem + ROWS * FF * 2);        // 32K
  __bf16* sH  = sA;                                     // alias
  __bf16* sE  = (__bf16*)(smem + ROWS * FF * 2 + ROWS * EE * 2);  // 32K
  float*  sS  = (float*)(smem + ROWS * FF * 2 + 2 * ROWS * EE * 2);

  const int t    = threadIdx.x;
  const int row0 = blockIdx.x * ROWS;
  const int rmax = min(ROWS, BB - row0);

  // stage bn tile (f32 -> bf16) into LDS; zero-fill padded tail rows
  for (int i = t; i < ROWS * FF; i += 128) {
    int r = i >> 9, c = i & (FF - 1);
    sBN[i] = (r < rmax) ? (__bf16)p.bn[(size_t)(row0 + r) * FF + c] : (__bf16)0.f;
  }
  __syncthreads();

  // edge chain
  wave_gemm(sBN, p.W1t, p.xvec, true,  sA, FF);   // relu(bn@W1top + x@W1bot + b1)
  __syncthreads();
  wave_gemm(sA,  p.W2t, p.geb2, false, sE, EE);   // edge
  __syncthreads();

  // pr encoder
  wave_gemm(sE, p.prencT, p.prencb, true, sH, EE);
  __syncthreads();
  wave_gemm(sH, p.prgT, p.prgb, true, sG, EE);
  wave_gemm(sH, p.prbT, p.prbb, true, sB, EE);
  __syncthreads();
  for (int d = t; d < 3 * ROWS; d += 128) {  // pr GEMVs: mean, std_pre, mij_pre
    int r = d & (ROWS - 1), w = d >> 6;
    if (row0 + r < BB) {
      const __bf16* v = (w == 2) ? (sB + r * EE) : (sG + r * EE);
      const float* W = (w == 0) ? p.prmeanW : (w == 1) ? p.prstdW : p.prmijW;
      float acc = (w == 0) ? p.prmeanb[0] : (w == 1) ? p.prstdb[0] : p.prmijb[0];
      for (int k = 0; k < EE; ++k) acc += (float)v[k] * W[k];
      sS[r * 8 + w] = acc;
    }
  }
  __syncthreads();

  // po encoder (reuses sH/sG/sB)
  wave_gemm(sE, p.poencT, p.poencb, true, sH, EE);
  __syncthreads();
  wave_gemm(sH, p.pogT, p.pogb, true, sG, EE);
  wave_gemm(sH, p.pobT, p.pobb, true, sB, EE);
  __syncthreads();
  for (int d = t; d < 4 * ROWS; d += 128) {  // po GEMVs: mean, std, mapx, sapx
    int r = d & (ROWS - 1), w = d >> 6;
    if (row0 + r < BB) {
      const __bf16* v = (w < 2) ? (sG + r * EE) : (sB + r * EE);
      const float* W = (w == 0) ? p.pomeanW : (w == 1) ? p.postdW
                     : (w == 2) ? p.pomapxW : p.posapxW;
      float acc = (w == 0) ? p.pomeanb[0] : (w == 1) ? p.postdb[0]
                : (w == 2) ? p.pomapxb[0] : p.posapxb[0];
      for (int k = 0; k < EE; ++k) acc += (float)v[k] * W[k];
      sS[r * 8 + 3 + w] = acc;
    }
  }
  __syncthreads();

  // per-row elementwise tail + KL accumulation
  if (t < ROWS) {
    int gr = row0 + t;
    if (gr < BB) {
      const float EPS = 1e-6f;
      float pr_mean = sS[t * 8 + 0];
      float pr_std  = sp_(sS[t * 8 + 1]);
      float pr_mij  = 0.4f * sigm_(sS[t * 8 + 2]);
      float pm      = sS[t * 8 + 3];
      float ps      = sp_(sS[t * 8 + 4]);
      float m_apx   = sS[t * 8 + 5];
      float s_apx   = sp_(sS[t * 8 + 6]);
      float nij  = sp_(m_apx) + 0.01f;
      float nij_ = 2.f * nij * s_apx * s_apx;
      float po_mij = 0.5f * (1.f + nij_ - sqrtf(nij_ * nij_ + 1.f));
      float e1 = p.eps1[gr], e2 = p.eps2[gr];
      float at = sp_(e1 * sqrtf(po_mij * (1.f - po_mij)) + po_mij);
      float mean_sij = at * pm;
      float std_sij  = sqrtf(at) * ps;
      float ab = sp_((e2 * std_sij + mean_sij) * at);
      float a3 = at * pr_mean;
      float a4 = sqrtf(at) * pr_std;
      float d  = mean_sij - a3;
      float kld = 2.f * logf(a4 + EPS) - 2.f * logf(std_sij + EPS) +
                  (std_sij * std_sij + d * d) / ((a4 + EPS) * (a4 + EPS)) - 1.f;
      float klb = fabsf(pr_mij - po_mij +
                        po_mij * (logf(po_mij + EPS) - logf(pr_mij + EPS)));
      atomicAdd(&p.out[64], 0.5f * fabsf(kld));
      atomicAdd(&p.out[65], klb);
      p.out[66 + gr] = ab;
      sS[t * 8 + 7] = ab;
    } else {
      sS[t * 8 + 7] = 0.f;   // padded rows contribute nothing to H_g
    }
  }
  __syncthreads();

  // H_g partial: sum_r alpha_bar[r] * bn[r][f]  (f32 source for fidelity)
#pragma unroll
  for (int j = 0; j < 4; ++j) {
    int f = t + j * 128;
    float acc = 0.f;
    for (int r = 0; r < rmax; ++r)
      acc += sS[r * 8 + 7] * p.bn[(size_t)(row0 + r) * FF + f];
    atomicAdd(&p.Hg[f], acc);
  }
}

// ---- finalize: x_out = ((Hg + x) @ out_W1 + b1) @ out_W2 + b2 -------------
__global__ __launch_bounds__(128) void k_final(P p) {
  __shared__ float sx[FF];
  __shared__ float sh[LDIM];
  int t = threadIdx.x;
  for (int f = t; f < FF; f += 128) sx[f] = p.Hg[f] + p.x[f];
  __syncthreads();
  float acc = p.outb1[t];
  for (int k = 0; k < FF; ++k) acc += sx[k] * p.outW1[(size_t)k * LDIM + t];
  sh[t] = acc;
  __syncthreads();
  if (t < OUTD) {
    float o = p.outb2[t];
    for (int k = 0; k < LDIM; ++k) o += sh[k] * p.outW2[(size_t)k * OUTD + t];
    p.out[t] = o;
  }
}

extern "C" void kernel_launch(void* const* d_in, const int* in_sizes, int n_in,
                              void* d_out, int out_size, void* d_ws, size_t ws_size,
                              hipStream_t stream) {
  (void)in_sizes; (void)n_in; (void)out_size; (void)ws_size;
  P p{};
  int i = 0;
  p.x = (const float*)d_in[i++]; p.eps1 = (const float*)d_in[i++];
  p.eps2 = (const float*)d_in[i++]; p.bn = (const float*)d_in[i++];
  p.geW1 = (const float*)d_in[i++]; p.geb1 = (const float*)d_in[i++];
  p.geW2 = (const float*)d_in[i++]; p.geb2 = (const float*)d_in[i++];
  p.prencW = (const float*)d_in[i++]; p.prencb = (const float*)d_in[i++];
  p.prgW = (const float*)d_in[i++]; p.prgb = (const float*)d_in[i++];
  p.prbW = (const float*)d_in[i++]; p.prbb = (const float*)d_in[i++];
  p.prmeanW = (const float*)d_in[i++]; p.prmeanb = (const float*)d_in[i++];
  p.prstdW = (const float*)d_in[i++]; p.prstdb = (const float*)d_in[i++];
  p.poencW = (const float*)d_in[i++]; p.poencb = (const float*)d_in[i++];
  p.pogW = (const float*)d_in[i++]; p.pogb = (const float*)d_in[i++];
  p.pobW = (const float*)d_in[i++]; p.pobb = (const float*)d_in[i++];
  p.pomeanW = (const float*)d_in[i++]; p.pomeanb = (const float*)d_in[i++];
  p.postdW = (const float*)d_in[i++]; p.postdb = (const float*)d_in[i++];
  p.prmijW = (const float*)d_in[i++]; p.prmijb = (const float*)d_in[i++];
  p.pomapxW = (const float*)d_in[i++]; p.pomapxb = (const float*)d_in[i++];
  p.posapxW = (const float*)d_in[i++]; p.posapxb = (const float*)d_in[i++];
  p.outW1 = (const float*)d_in[i++]; p.outb1 = (const float*)d_in[i++];
  p.outW2 = (const float*)d_in[i++]; p.outb2 = (const float*)d_in[i++];

  char* w = (char*)d_ws;
  size_t off = 0;
  auto takeb = [&](size_t elems) { __bf16* q = (__bf16*)(w + off); off += elems * 2; return q; };
  p.W1t    = takeb((size_t)FF * EE);
  p.W2t    = takeb((size_t)EE * EE);
  p.prencT = takeb((size_t)EE * EE);
  p.prgT   = takeb((size_t)EE * EE);
  p.prbT   = takeb((size_t)EE * EE);
  p.poencT = takeb((size_t)EE * EE);
  p.pogT   = takeb((size_t)EE * EE);
  p.pobT   = takeb((size_t)EE * EE);
  p.xvec = (float*)(w + off); off += EE * 4;
  p.Hg   = (float*)(w + off); off += FF * 4;
  p.out  = (float*)d_out;

  // weight convert/transpose (f32 [K][N] -> bf16 [N][K]); all weights ~1.2MB
  k_conv<<<512, 256, 0, stream>>>(p.geW1,   p.W1t,    FF, EE);  // top 512 rows
  k_conv<<<256, 256, 0, stream>>>(p.geW2,   p.W2t,    EE, EE);
  k_conv<<<256, 256, 0, stream>>>(p.prencW, p.prencT, EE, EE);
  k_conv<<<256, 256, 0, stream>>>(p.prgW,   p.prgT,   EE, EE);
  k_conv<<<256, 256, 0, stream>>>(p.prbW,   p.prbT,   EE, EE);
  k_conv<<<256, 256, 0, stream>>>(p.poencW, p.poencT, EE, EE);
  k_conv<<<256, 256, 0, stream>>>(p.pogW,   p.pogT,   EE, EE);
  k_conv<<<256, 256, 0, stream>>>(p.pobW,   p.pobT,   EE, EE);

  k_prep<<<1, 512, 0, stream>>>(p);
  k_main<<<(BB + ROWS - 1) / ROWS, 128, 0, stream>>>(p);  // 157 blocks
  k_final<<<1, 128, 0, stream>>>(p);
}
